// HelmholtzLoss_29970281792059
// MI455X (gfx1250) — matrix operand order repaired and checked by reading, not compiled, not run on hardware
//
#include <hip/hip_runtime.h>
#include <math.h>

typedef __attribute__((ext_vector_type(16))) _Float16 v16h;
typedef __attribute__((ext_vector_type(8)))  _Float16 v8h_t;
typedef __attribute__((ext_vector_type(8)))  float    v8f;

#define NF      16
#define HID     128
#define NPTS    131072
#define CSOUND  343.0f

#define WAVES_PER_WG 8
#define THREADS      (WAVES_PER_WG * 32)
#define NTILES       (NPTS / 16)                // 8192
#define NBLOCKS      (NTILES / WAVES_PER_WG)    // 1024
#define PAD          136                        // halves per row: 272B = 16B-aligned, bank-spread

// dynamic LDS layout (all offsets 16B aligned):
//  w1l   : 3*128 f32
//  b1l   : 128 f32
//  b2l   : 128 f32
//  w2t   : [128 n][PAD k] f16   (W2 transposed)
//  w3t   : [32  n][PAD k] f16   (W3 transposed)
//  stg   : per-wave 2 x [16 m][PAD k] f16  (a2, lapA2 staging)
#define SMEM_F32   (3 * 128 + 128 + 128)
#define SMEM_HALF  (128 * PAD + 32 * PAD + WAVES_PER_WG * 2 * 16 * PAD)
#define SMEM_BYTES (SMEM_F32 * 4 + SMEM_HALF * 2)

static __device__ inline v8f wmma16(v16h a, v16h b, v8f c) {
  return __builtin_amdgcn_wmma_f32_16x16x32_f16(
      /*neg_a=*/false, a, /*neg_b=*/false, b,
      /*c_mod=*/(short)0, c, /*reuse_a=*/false, /*reuse_b=*/false);
}

// Branchless tanh on the hardware TRANS path:
//   tanh(x) = 1 - 2/(exp(2x)+1),  exp(2x) = 2^(2*log2(e)*x) via v_exp_f32 (exp2).
// Saturates correctly at +-1 through inf/zero, no EXEC divergence.
static __device__ inline float fast_tanh(float x) {
  const float e = __builtin_amdgcn_exp2f(x * 2.88539008177793f); // 2*log2(e)
  return 1.0f - 2.0f * __builtin_amdgcn_rcpf(e + 1.0f);
}

// B fragment (32x16 f16): elems 0-7 = k..k+7, 8-15 = k+16..k+23 for this lane's (half, n)
static __device__ inline v16h load_bfrag(const _Float16* p) {
  v8h_t lo = *(const v8h_t*)p;
  v8h_t hi = *(const v8h_t*)(p + 16);
  v16h r;
#pragma unroll
  for (int e = 0; e < 8; ++e) { r[e] = lo[e]; r[8 + e] = hi[e]; }
  return r;
}

// A fragment (16x32 f16) from row-major [m][k] staging: elems 0-7 = k..k+7, 8-15 = k+16..k+23
static __device__ inline v16h load_afrag(const _Float16* p) {
  v8h_t lo = *(const v8h_t*)p;
  v8h_t hi = *(const v8h_t*)(p + 16);
  v16h r;
#pragma unroll
  for (int e = 0; e < 8; ++e) { r[e] = lo[e]; r[8 + e] = hi[e]; }
  return r;
}

extern "C" __global__ __launch_bounds__(THREADS) void helm_loss_kernel(
    const float* __restrict__ inp,    // [N,3]
    const float* __restrict__ omega,  // [16]
    const float* __restrict__ W1,     // [3,128]
    const float* __restrict__ b1,     // [128]
    const float* __restrict__ W2,     // [128,128]
    const float* __restrict__ b2,     // [128]
    const float* __restrict__ W3,     // [128,32]
    const float* __restrict__ b3,     // [32]
    float* __restrict__ partial)      // [NBLOCKS]
{
  extern __shared__ char smem[];
  float*    w1l = (float*)smem;                       // 3*128
  float*    b1l = w1l + 3 * 128;                      // 128
  float*    b2l = b1l + 128;                          // 128
  _Float16* w2t = (_Float16*)(b2l + 128);             // 128*PAD
  _Float16* w3t = w2t + 128 * PAD;                    // 32*PAD
  _Float16* stg = w3t + 32 * PAD;                     // per-wave staging
  __shared__ float red[THREADS];

  const int tid  = threadIdx.x;
  const int wid  = tid >> 5;
  const int lane = tid & 31;
  const int m    = lane & 15;   // point row within tile / column within B tile
  const int half = lane >> 4;   // lane-half selects K sub-range

  // ---- stage weights into LDS (f16, transposed for B-fragment reads) ----
  for (int i = tid; i < 3 * 128; i += THREADS) w1l[i] = W1[i];
  for (int i = tid; i < 128; i += THREADS) { b1l[i] = b1[i]; b2l[i] = b2[i]; }
  for (int i = tid; i < 128 * 128; i += THREADS) {
    int j = i >> 7, n = i & 127;                      // W2[j][n]
    w2t[n * PAD + j] = (_Float16)W2[i];
  }
  for (int i = tid; i < 128 * 32; i += THREADS) {
    int j = i >> 5, n = i & 31;                       // W3[j][n]
    w3t[n * PAD + j] = (_Float16)W3[i];
  }
  __syncthreads();

  const int tile = blockIdx.x * WAVES_PER_WG + wid;
  const int p    = tile * 16 + m;
  const float x0 = inp[3 * p + 0];
  const float x1 = inp[3 * p + 1];
  const float x2 = inp[3 * p + 2];

  // ---- layer 1, built directly in WMMA A-fragment layout ----
  // 5 A matrices [16 x 128]: a1, g1*W1[d,:] (d=0..2), -2*a1*g1*sum_d W1[d,:]^2
  v16h A1[4], P0[4], P1[4], P2[4], SS[4];
#pragma unroll
  for (int c = 0; c < 4; ++c) {
#pragma unroll
    for (int e = 0; e < 16; ++e) {
      const int k  = 32 * c + half * 8 + (e < 8 ? e : 8 + e);
      const float wa = w1l[k], wb = w1l[128 + k], wc = w1l[256 + k];
      const float z  = x0 * wa + x1 * wb + x2 * wc + b1l[k];
      const float a  = fast_tanh(z);
      const float g  = 1.0f - a * a;
      A1[c][e] = (_Float16)a;
      P0[c][e] = (_Float16)(g * wa);
      P1[c][e] = (_Float16)(g * wb);
      P2[c][e] = (_Float16)(g * wc);
      SS[c][e] = (_Float16)(-2.0f * a * g * (wa * wa + wb * wb + wc * wc));
    }
  }

  _Float16* stA = stg + wid * (2 * 16 * PAD);
  _Float16* stL = stA + 16 * PAD;

  // ---- layer 2: five [16x128] @ W2[128x128] via v_wmma_f32_16x16x32_f16 ----
  for (int nt = 0; nt < 8; ++nt) {
    v8f accZ = {}, accT0 = {}, accT1 = {}, accT2 = {}, accS = {};
    const int n = nt * 16 + m;
#pragma unroll
    for (int c = 0; c < 4; ++c) {
      const v16h B = load_bfrag(w2t + n * PAD + c * 32 + half * 8);
      accZ  = wmma16(A1[c], B, accZ);
      accT0 = wmma16(P0[c], B, accT0);
      accT1 = wmma16(P1[c], B, accT1);
      accT2 = wmma16(P2[c], B, accT2);
      accS  = wmma16(SS[c], B, accS);
    }
    // elementwise: a2, g2, lapA2; stage to LDS in [m][k] layout for layer-3 A frags
    const float bias2 = b2l[n];
#pragma unroll
    for (int r = 0; r < 8; ++r) {
      const float z2 = accZ[r] + bias2;
      const float a2 = fast_tanh(z2);
      const float g2 = 1.0f - a2 * a2;
      const float q  = accT0[r] * accT0[r] + accT1[r] * accT1[r] + accT2[r] * accT2[r];
      const float la = g2 * accS[r] - 2.0f * a2 * g2 * q;
      const int   mm = r + half * 8;       // C-layout row
      stA[mm * PAD + n] = (_Float16)a2;
      stL[mm * PAD + n] = (_Float16)la;
    }
  }
  __syncthreads();   // uniform; also guarantees DS store->load ordering

  // ---- layer 3: [16x128] @ W3[128x32] for u and for the Laplacian ----
  v16h Au[4], Al[4];
#pragma unroll
  for (int c = 0; c < 4; ++c) {
    Au[c] = load_afrag(stA + m * PAD + c * 32 + half * 8);
    Al[c] = load_afrag(stL + m * PAD + c * 32 + half * 8);
  }
  v8f accUre = {}, accUim = {}, accLre = {}, accLim = {};
#pragma unroll
  for (int c = 0; c < 4; ++c) {
    const v16h B0 = load_bfrag(w3t + (m)      * PAD + c * 32 + half * 8); // real cols 0..15
    const v16h B1 = load_bfrag(w3t + (16 + m) * PAD + c * 32 + half * 8); // imag cols 16..31
    accUre = wmma16(Au[c], B0, accUre);
    accLre = wmma16(Al[c], B0, accLre);
    accUim = wmma16(Au[c], B1, accUim);
    accLim = wmma16(Al[c], B1, accLim);
  }

  // ---- Helmholtz residual: lap + (omega_f/c)^2 * u, skip channel 0 ----
  const int   f   = m;                 // this lane's channel (C layout: N = lane%16)
  const float om  = omega[f];
  const float k2  = (om / CSOUND) * (om / CSOUND);
  const float bre = b3[f];
  const float bim = b3[16 + f];
  float s = 0.0f;
#pragma unroll
  for (int r = 0; r < 8; ++r) {
    const float ur = accUre[r] + bre;
    const float ui = accUim[r] + bim;
    const float rr = accLre[r] + k2 * ur;
    const float ri = accLim[r] + k2 * ui;
    s += rr * rr + ri * ri;
  }
  if (f == 0) s = 0.0f;

  // deterministic block reduction
  red[tid] = s;
  __syncthreads();
  if (tid == 0) {
    float t = 0.0f;
    for (int i = 0; i < THREADS; ++i) t += red[i];
    partial[blockIdx.x] = t;
  }
}

extern "C" __global__ __launch_bounds__(256) void helm_reduce_kernel(
    const float* __restrict__ partial, float* __restrict__ out, int nb) {
  __shared__ float red[256];
  float s = 0.0f;
  for (int i = threadIdx.x; i < nb; i += 256) s += partial[i];
  red[threadIdx.x] = s;
  __syncthreads();
  if (threadIdx.x == 0) {
    float t = 0.0f;
    for (int i = 0; i < 256; ++i) t += red[i];
    out[0] = t / (float)(NPTS * (NF - 1));   // mean of |residual|^2 over [N, F-1]
  }
}

extern "C" void kernel_launch(void* const* d_in, const int* in_sizes, int n_in,
                              void* d_out, int out_size, void* d_ws, size_t ws_size,
                              hipStream_t stream) {
  (void)in_sizes; (void)n_in; (void)out_size; (void)ws_size;
  const float* inp   = (const float*)d_in[0];
  const float* omega = (const float*)d_in[1];
  const float* W1    = (const float*)d_in[2];
  const float* b1    = (const float*)d_in[3];
  const float* W2    = (const float*)d_in[4];
  const float* b2    = (const float*)d_in[5];
  const float* W3    = (const float*)d_in[6];
  const float* b3    = (const float*)d_in[7];
  float* partial = (float*)d_ws;
  float* out     = (float*)d_out;

  helm_loss_kernel<<<NBLOCKS, THREADS, SMEM_BYTES, stream>>>(
      inp, omega, W1, b1, W2, b2, W3, b3, partial);
  helm_reduce_kernel<<<1, 256, 0, stream>>>(partial, out, NBLOCKS);
}
